// GraphUNet_15161234555391
// MI455X (gfx1250) — compile-verified, matrix operand order checked by reading, not run on hardware
//
#include <hip/hip_runtime.h>
#include <hip/hip_bf16.h>
#include <math.h>

typedef __attribute__((ext_vector_type(16))) _Float16 v16h;
typedef __attribute__((ext_vector_type(8)))  float    v8f;
typedef __attribute__((ext_vector_type(8)))  unsigned v8u;

#define N0   100000
#define EE   1600000
#define CC   64
#define FIN  128
#define NCLS 16

// ---- CDNA5 async global->LDS path (guarded; falls back to plain LDS stores) ----
#if defined(__gfx1250__) && __has_builtin(__builtin_amdgcn_global_load_async_to_lds_b128) && __has_builtin(__builtin_amdgcn_s_wait_asynccnt)
#define USE_ASYNC_LDS 1
typedef __attribute__((ext_vector_type(4))) int v4i;
typedef __attribute__((address_space(1))) v4i g_v4i;   // global int4
typedef __attribute__((address_space(3))) v4i l_v4i;   // LDS int4
#else
#define USE_ASYNC_LDS 0
#endif

static inline int nblk(long long n, int b) { return (int)((n + b - 1) / b); }

// ---------------------------------------------------------------------------
// fills
// ---------------------------------------------------------------------------
__global__ void k_fill_f32(float* p, float v, long long n) {
  long long i = (long long)blockIdx.x * blockDim.x + threadIdx.x;
  if (i < n) p[i] = v;
}
__global__ void k_fill_u32(unsigned* p, unsigned v, long long n) {
  long long i = (long long)blockIdx.x * blockDim.x + threadIdx.x;
  if (i < n) p[i] = v;
}
__global__ void k_fill_i32(int* p, int v, long long n) {
  long long i = (long long)blockIdx.x * blockDim.x + threadIdx.x;
  if (i < n) p[i] = v;
}
__global__ void k_copy_f32(float* __restrict__ dst, const float* __restrict__ src, long long n) {
  long long i = (long long)blockIdx.x * blockDim.x + threadIdx.x;
  if (i < n) dst[i] = src[i];
}

// ---------------------------------------------------------------------------
// WMMA GEMM:  C[M x Nn] = A[M x K] * B[K x Nn], row-major f32 in/out,
// f16 multiply, f32 accumulate.  K % 32 == 0, Nn == NT*16.
// Block: 256 threads = 8 waves -> 128 rows x Nn cols per block.
// Each wave: 16 rows x Nn cols = NT accumulators, NT wmma per K-step.
// A tile staged raw f32 via async-to-LDS; B tile staged as packed f16 pairs.
// ---------------------------------------------------------------------------
template <int NT>
__global__ void __launch_bounds__(256)
k_gemm_wmma(const float* __restrict__ A, const float* __restrict__ B,
            float* __restrict__ C, int M, int K) {
  constexpr int BN = NT * 16;
  __shared__ float    Atile[128 * 32];      // 16 KB, [row][k] raw f32
  __shared__ unsigned Bpair[64 * 16];       // 4 KB,  [col][kpair] packed 2xf16

  const int tid  = threadIdx.x;
  const int lane = tid & 31;
  const int w    = tid >> 5;                // wave 0..7
  const int half = lane >> 4;               // K-half select (ISA 7.12.2)
  const int r    = lane & 15;               // row (A) / col (B) within tile
  const int mb   = blockIdx.x * 128;
  const int Nn   = BN;

  v8f acc[NT];
#pragma unroll
  for (int t = 0; t < NT; ++t) acc[t] = v8f{};

  const bool waveActive = (mb + w * 16) < M;   // uniform per wave

  for (int kk = 0; kk < K; kk += 32) {
    // ---- stage A tile: 128 rows x 32 f32 = 1024 x 16B chunks, 4 per thread
#pragma unroll
    for (int i = 0; i < 4; ++i) {
      const int c    = i * 256 + tid;       // chunk id
      const int row  = c >> 3;              // 0..127
      const int part = c & 7;               // 0..7 (4 floats each)
      const int grow = mb + row;
      if (grow < M) {
#if USE_ASYNC_LDS
        __builtin_amdgcn_global_load_async_to_lds_b128(
            (g_v4i*)(A + (size_t)grow * K + kk + part * 4),
            (l_v4i*)&Atile[row * 32 + part * 4], 0, 0);
#else
        *(float4*)&Atile[row * 32 + part * 4] =
            *(const float4*)&A[(size_t)grow * K + kk + part * 4];
#endif
      } else {
        *(float4*)&Atile[row * 32 + part * 4] = float4{0.f, 0.f, 0.f, 0.f};
      }
    }
    // ---- stage B tile: 16 k-pairs x BN cols, packed f16 pairs
#pragma unroll
    for (int i = 0; i < (BN * 16) / 256; ++i) {
      const int p   = i * 256 + tid;        // pair id
      const int col = p >> 4;
      const int pr  = p & 15;               // k-pair 0..15
      const int k0  = kk + pr * 2;
      const _Float16 h0 = (_Float16)B[(size_t)k0 * Nn + col];
      const _Float16 h1 = (_Float16)B[(size_t)(k0 + 1) * Nn + col];
      const unsigned u = (unsigned)__builtin_bit_cast(unsigned short, h0) |
                         ((unsigned)__builtin_bit_cast(unsigned short, h1) << 16);
      Bpair[col * 16 + pr] = u;
    }
#if USE_ASYNC_LDS
    __builtin_amdgcn_s_wait_asynccnt(0);
#endif
    __syncthreads();

    if (waveActive) {
      // A fragment: row = w*16 + r; K pattern per ISA 7.12.2 (f16 A 16x32)
      const float* arow = &Atile[(w * 16 + r) * 32];
      const float4 q0 = *(const float4*)(arow + half * 8);
      const float4 q1 = *(const float4*)(arow + half * 8 + 4);
      const float4 q2 = *(const float4*)(arow + 16 + half * 8);
      const float4 q3 = *(const float4*)(arow + 16 + half * 8 + 4);
      v16h av;
      av[0]  = (_Float16)q0.x; av[1]  = (_Float16)q0.y;
      av[2]  = (_Float16)q0.z; av[3]  = (_Float16)q0.w;
      av[4]  = (_Float16)q1.x; av[5]  = (_Float16)q1.y;
      av[6]  = (_Float16)q1.z; av[7]  = (_Float16)q1.w;
      av[8]  = (_Float16)q2.x; av[9]  = (_Float16)q2.y;
      av[10] = (_Float16)q2.z; av[11] = (_Float16)q2.w;
      av[12] = (_Float16)q3.x; av[13] = (_Float16)q3.y;
      av[14] = (_Float16)q3.z; av[15] = (_Float16)q3.w;
#pragma unroll
      for (int t = 0; t < NT; ++t) {
        const unsigned* bcol = &Bpair[(t * 16 + r) * 16];
        const uint4 lo = *(const uint4*)(bcol + half * 4);
        const uint4 hi = *(const uint4*)(bcol + 8 + half * 4);
        v8u bu;
        bu[0] = lo.x; bu[1] = lo.y; bu[2] = lo.z; bu[3] = lo.w;
        bu[4] = hi.x; bu[5] = hi.y; bu[6] = hi.z; bu[7] = hi.w;
        const v16h bv = __builtin_bit_cast(v16h, bu);
        acc[t] = __builtin_amdgcn_wmma_f32_16x16x32_f16(false, av, false, bv,
                                                        (short)0, acc[t], false, false);
      }
    }
    __syncthreads();
  }

#pragma unroll
  for (int t = 0; t < NT; ++t) {
#pragma unroll
    for (int v = 0; v < 8; ++v) {
      const int rr = mb + w * 16 + v + half * 8;   // C/D layout (ISA 7.12.2)
      if (rr < M) C[(size_t)rr * Nn + t * 16 + r] = acc[t][v];
    }
  }
}

// ---------------------------------------------------------------------------
// GCN pieces
// ---------------------------------------------------------------------------
__global__ void k_deg(const int* __restrict__ col, const float* __restrict__ w,
                      float* __restrict__ deg, int ne) {
  int e = blockIdx.x * blockDim.x + threadIdx.x;
  if (e >= ne) return;
  float ww = w ? w[e] : 1.0f;
  if (ww != 0.f) atomicAdd(&deg[col[e]], ww);
}
__global__ void k_dinv(const float* __restrict__ deg, float* __restrict__ dinv, int n) {
  int i = blockIdx.x * blockDim.x + threadIdx.x;
  if (i >= n) return;
  float d = deg[i];
  dinv[i] = d > 0.f ? rsqrtf(d) : 0.f;
}
// one thread per (edge, feature-pair)
__global__ void k_agg(const int* __restrict__ row, const int* __restrict__ col,
                      const float* __restrict__ w, const float* __restrict__ dinv,
                      const float* __restrict__ xw, float* __restrict__ agg,
                      int F, long long total) {
  long long tid = (long long)blockIdx.x * blockDim.x + threadIdx.x;
  if (tid >= total) return;
  const int fh = F >> 1;
  const int e  = (int)(tid / fh);
  const int f  = (int)(tid % fh) * 2;
  float ww = w ? w[e] : 1.0f;
  if (ww == 0.f) return;
  const int rs = row[e], cd = col[e];
  const float coef = ww * dinv[rs] * dinv[cd];
  __builtin_prefetch(&agg[(size_t)cd * F + f], 1, 1);
  const float2 xv = *(const float2*)(xw + (size_t)rs * F + f);
  atomicAdd(&agg[(size_t)cd * F + f],     xv.x * coef);
  atomicAdd(&agg[(size_t)cd * F + f + 1], xv.y * coef);
}
__global__ void k_finish(const float* __restrict__ agg, const float* __restrict__ xw,
                         const float* __restrict__ dinv, const float* __restrict__ bias,
                         float* __restrict__ out, int n, int F, int relu) {
  long long tid = (long long)blockIdx.x * blockDim.x + threadIdx.x;
  if (tid >= (long long)n * F) return;
  const int i = (int)(tid / F);
  const int f = (int)(tid % F);
  const float di = dinv[i];
  float v = agg[tid] + 2.f * di * di * xw[tid] + bias[f];
  if (relu) v = fmaxf(v, 0.f);
  out[tid] = v;
}

// ---------------------------------------------------------------------------
// TopK pooling
// ---------------------------------------------------------------------------
__global__ void k_score(const float* __restrict__ x, const float* __restrict__ p,
                        float* __restrict__ score, unsigned* __restrict__ keys,
                        int n, int F) {
  int i = blockIdx.x * blockDim.x + threadIdx.x;
  if (i >= n) return;
  float s = 0.f, pn = 0.f;
  for (int c = 0; c < F; ++c) {
    float pc = p[c];
    s  += x[(size_t)i * F + c] * pc;
    pn += pc * pc;
  }
  float sc = s * rsqrtf(pn);
  score[i] = sc;
  unsigned u = __float_as_uint(sc);
  keys[i] = (u & 0x80000000u) ? ~u : (u | 0x80000000u);   // order-preserving
}
__global__ void k_hist_hi(const unsigned* __restrict__ keys, unsigned* __restrict__ hist, int n) {
  int i = blockIdx.x * blockDim.x + threadIdx.x;
  if (i < n) atomicAdd(&hist[keys[i] >> 16], 1u);
}
__global__ void k_find_hi(const unsigned* __restrict__ hist, unsigned* __restrict__ sel, int k) {
  unsigned acc = 0;
  for (int b = 65535; b >= 0; --b) {
    unsigned h = hist[b];
    if (acc + h >= (unsigned)k) { sel[0] = (unsigned)b; sel[1] = acc; return; }
    acc += h;
  }
  sel[0] = 0; sel[1] = acc;
}
__global__ void k_hist_lo(const unsigned* __restrict__ keys, const unsigned* __restrict__ sel,
                          unsigned* __restrict__ hist, int n) {
  int i = blockIdx.x * blockDim.x + threadIdx.x;
  if (i >= n) return;
  unsigned kk = keys[i];
  if ((kk >> 16) == sel[0]) atomicAdd(&hist[kk & 0xFFFFu], 1u);
}
__global__ void k_find_lo(const unsigned* __restrict__ hist, unsigned* __restrict__ sel, int k) {
  const unsigned bt = sel[0], above = sel[1];
  unsigned acc = 0;
  for (int l = 65535; l >= 0; --l) {
    unsigned h = hist[l];
    if (above + acc + h >= (unsigned)k) {
      sel[2] = (bt << 16) | (unsigned)l;       // threshold key T
      sel[3] = above + acc;                    // strictly greater than T
      sel[4] = (unsigned)k - (above + acc);    // ties needed at == T
      return;
    }
    acc += h;
  }
  sel[2] = 0; sel[3] = above + acc; sel[4] = 0;
}
// per-block exclusive scan of packed (above-count | tie-count<<32)
__global__ void k_flag_scan(const unsigned* __restrict__ keys, const unsigned* __restrict__ sel,
                            unsigned long long* __restrict__ prefix,
                            unsigned long long* __restrict__ sums, int n) {
  __shared__ unsigned long long sh[256];
  const int i = blockIdx.x * 256 + threadIdx.x;
  const unsigned T = sel[2];
  unsigned long long v = 0;
  if (i < n) {
    unsigned kk = keys[i];
    if (kk > T)       v = 1ULL;
    else if (kk == T) v = 1ULL << 32;
  }
  sh[threadIdx.x] = v;
  __syncthreads();
  for (int off = 1; off < 256; off <<= 1) {
    unsigned long long t = 0;
    if ((int)threadIdx.x >= off) t = sh[threadIdx.x - off];
    __syncthreads();
    sh[threadIdx.x] += t;
    __syncthreads();
  }
  unsigned long long incl = sh[threadIdx.x];
  if (i < n) prefix[i] = incl - v;                 // exclusive
  if (threadIdx.x == 255) sums[blockIdx.x] = incl; // block total
}
__global__ void k_scan_sums(unsigned long long* sums, int nb) {
  unsigned long long run = 0;
  for (int b = 0; b < nb; ++b) {
    unsigned long long t = sums[b];
    sums[b] = run;
    run += t;
  }
}
__global__ void k_compact(const unsigned* __restrict__ keys, const float* __restrict__ score,
                          const unsigned* __restrict__ sel,
                          const unsigned long long* __restrict__ prefix,
                          const unsigned long long* __restrict__ sums,
                          int* __restrict__ perm, float* __restrict__ topv, int n) {
  int i = blockIdx.x * blockDim.x + threadIdx.x;
  if (i >= n) return;
  const unsigned T = sel[2], nAbove = sel[3], need = sel[4];
  const unsigned kk = keys[i];
  const unsigned long long base = prefix[i] + sums[i >> 8];
  const unsigned aPre = (unsigned)(base & 0xFFFFFFFFull);
  const unsigned tPre = (unsigned)(base >> 32);
  int pos = -1;
  if (kk > T)                      pos = (int)aPre;
  else if (kk == T && tPre < need) pos = (int)(nAbove + tPre);
  if (pos >= 0) { perm[pos] = i; topv[pos] = score[i]; }
}
__global__ void k_pool_x(const float* __restrict__ x, const int* __restrict__ perm,
                         const float* __restrict__ topv, float* __restrict__ xp,
                         int k, int F) {
  long long tid = (long long)blockIdx.x * blockDim.x + threadIdx.x;
  if (tid >= (long long)k * F) return;
  const int j = (int)(tid / F);
  const int c = (int)(tid % F);
  xp[tid] = x[(size_t)perm[j] * F + c] * tanhf(topv[j]);
}
__global__ void k_map_set(const int* __restrict__ perm, int* __restrict__ mapping, int k) {
  int j = blockIdx.x * blockDim.x + threadIdx.x;
  if (j < k) mapping[perm[j]] = j;
}
__global__ void k_relabel(const int* __restrict__ rowIn, const int* __restrict__ colIn,
                          const float* __restrict__ wIn, const int* __restrict__ mapping,
                          int* __restrict__ rowOut, int* __restrict__ colOut,
                          float* __restrict__ wOut, int ne) {
  int e = blockIdx.x * blockDim.x + threadIdx.x;
  if (e >= ne) return;
  const int nr = mapping[rowIn[e]];
  const int nc = mapping[colIn[e]];
  const bool valid = (nr >= 0) && (nc >= 0);
  const float ww = wIn ? wIn[e] : 1.0f;
  rowOut[e] = valid ? nr : 0;
  colOut[e] = valid ? nc : 0;
  wOut[e]   = valid ? ww : 0.f;
}
// unpool scatter-add (perm entries are unique -> plain add)
__global__ void k_scatter_add(float* __restrict__ out, const float* __restrict__ x,
                              const int* __restrict__ perm, int k, int F) {
  long long tid = (long long)blockIdx.x * blockDim.x + threadIdx.x;
  if (tid >= (long long)k * F) return;
  const int j = (int)(tid / F);
  const int c = (int)(tid % F);
  out[(size_t)perm[j] * F + c] += x[tid];
}
__global__ void k_logsoftmax(float* __restrict__ o, int n) {
  int i = blockIdx.x * blockDim.x + threadIdx.x;
  if (i >= n) return;
  float* rowp = o + (size_t)i * NCLS;
  float m = rowp[0];
  for (int c = 1; c < NCLS; ++c) m = fmaxf(m, rowp[c]);
  float s = 0.f;
  for (int c = 0; c < NCLS; ++c) s += __expf(rowp[c] - m);
  const float ls = __logf(s);
  for (int c = 0; c < NCLS; ++c) rowp[c] = rowp[c] - m - ls;
}

// ---------------------------------------------------------------------------
// host orchestration
// ---------------------------------------------------------------------------
struct Scratch {
  float *deg, *dinv, *xw, *agg;
  float *score, *topv;
  unsigned *keys, *hist, *sel;
  unsigned long long *prefix, *sums;
  int *mapping;
};

static void run_gcn(const float* x, const int* row, const int* col, const float* w,
                    const float* Wm, const float* bias, int n, int K, int F,
                    float* out, int relu, const Scratch& S, hipStream_t s) {
  k_fill_f32<<<nblk(n, 256), 256, 0, s>>>(S.deg, 2.0f, n);
  k_deg<<<nblk(EE, 256), 256, 0, s>>>(col, w, S.deg, EE);
  k_dinv<<<nblk(n, 256), 256, 0, s>>>(S.deg, S.dinv, n);
  {
    const int blocks = (n + 127) / 128;
    if (F == 64) k_gemm_wmma<4><<<blocks, 256, 0, s>>>(x, Wm, S.xw, n, K);
    else         k_gemm_wmma<1><<<blocks, 256, 0, s>>>(x, Wm, S.xw, n, K);
  }
  k_fill_f32<<<nblk((long long)n * F, 256), 256, 0, s>>>(S.agg, 0.f, (long long)n * F);
  {
    long long total = (long long)EE * (F / 2);
    k_agg<<<nblk(total, 256), 256, 0, s>>>(row, col, w, S.dinv, S.xw, S.agg, F, total);
  }
  k_finish<<<nblk((long long)n * F, 256), 256, 0, s>>>(S.agg, S.xw, S.dinv, bias, out, n, F, relu);
}

static void run_pool(const float* x, const float* p, int n, int k,
                     const int* rowIn, const int* colIn, const float* wIn,
                     int* rowOut, int* colOut, float* wOut,
                     int* perm, float* xp, const Scratch& S, hipStream_t s) {
  k_score<<<nblk(n, 256), 256, 0, s>>>(x, p, S.score, S.keys, n, CC);
  k_fill_u32<<<nblk(131072, 256), 256, 0, s>>>(S.hist, 0u, 131072);
  k_hist_hi<<<nblk(n, 256), 256, 0, s>>>(S.keys, S.hist, n);
  k_find_hi<<<1, 1, 0, s>>>(S.hist, S.sel, k);
  k_hist_lo<<<nblk(n, 256), 256, 0, s>>>(S.keys, S.sel, S.hist + 65536, n);
  k_find_lo<<<1, 1, 0, s>>>(S.hist + 65536, S.sel, k);
  const int nb = nblk(n, 256);
  k_flag_scan<<<nb, 256, 0, s>>>(S.keys, S.sel, S.prefix, S.sums, n);
  k_scan_sums<<<1, 1, 0, s>>>(S.sums, nb);
  k_compact<<<nblk(n, 256), 256, 0, s>>>(S.keys, S.score, S.sel, S.prefix, S.sums,
                                         perm, S.topv, n);
  k_pool_x<<<nblk((long long)k * CC, 256), 256, 0, s>>>(x, perm, S.topv, xp, k, CC);
  k_fill_i32<<<nblk(n, 256), 256, 0, s>>>(S.mapping, -1, n);
  k_map_set<<<nblk(k, 256), 256, 0, s>>>(perm, S.mapping, k);
  k_relabel<<<nblk(EE, 256), 256, 0, s>>>(rowIn, colIn, wIn, S.mapping,
                                          rowOut, colOut, wOut, EE);
}

extern "C" void kernel_launch(void* const* d_in, const int* in_sizes, int n_in,
                              void* d_out, int out_size, void* d_ws, size_t ws_size,
                              hipStream_t stream) {
  (void)in_sizes; (void)n_in; (void)out_size; (void)ws_size;
  const int KS[3] = {50000, 25000, 12500};

  const float* x_in  = (const float*)d_in[0];
  const int*   row0  = (const int*)d_in[1];
  const int*   col0  = row0 + EE;
  const float* Wenc[4] = {(const float*)d_in[2], (const float*)d_in[4],
                          (const float*)d_in[6], (const float*)d_in[8]};
  const float* benc[4] = {(const float*)d_in[3], (const float*)d_in[5],
                          (const float*)d_in[7], (const float*)d_in[9]};
  const float* ppool[3] = {(const float*)d_in[10], (const float*)d_in[11],
                           (const float*)d_in[12]};
  const float* Wdec[3] = {(const float*)d_in[13], (const float*)d_in[15],
                          (const float*)d_in[17]};
  const float* bdec[3] = {(const float*)d_in[14], (const float*)d_in[16],
                          (const float*)d_in[18]};

  // bump allocator over d_ws
  char* wsp = (char*)d_ws;
  size_t off = 0;
  auto alloc = [&](size_t bytes) -> void* {
    off = (off + 255) & ~(size_t)255;
    void* p = wsp + off;
    off += bytes;
    return p;
  };

  int*   rowL[4];  int* colL[4];  float* wL[4];
  rowL[0] = const_cast<int*>(row0); colL[0] = const_cast<int*>(col0); wL[0] = nullptr;
  for (int l = 1; l <= 3; ++l) {
    rowL[l] = (int*)alloc(sizeof(int) * EE);
    colL[l] = (int*)alloc(sizeof(int) * EE);
    wL[l]   = (float*)alloc(sizeof(float) * EE);
  }
  float* xlev[4];
  xlev[0] = (float*)alloc(sizeof(float) * (size_t)N0 * CC);
  xlev[1] = (float*)alloc(sizeof(float) * (size_t)KS[0] * CC);
  xlev[2] = (float*)alloc(sizeof(float) * (size_t)KS[1] * CC);
  xlev[3] = (float*)alloc(sizeof(float) * (size_t)KS[2] * CC);
  float* tmp = (float*)alloc(sizeof(float) * (size_t)N0 * CC);  // pooled-x / combined-x
  float* xd  = (float*)alloc(sizeof(float) * (size_t)N0 * CC);  // decoder activations

  Scratch S;
  S.deg   = (float*)alloc(sizeof(float) * N0);
  S.dinv  = (float*)alloc(sizeof(float) * N0);
  S.xw    = (float*)alloc(sizeof(float) * (size_t)N0 * CC);
  S.agg   = (float*)alloc(sizeof(float) * (size_t)N0 * CC);
  S.score = (float*)alloc(sizeof(float) * N0);
  S.topv  = (float*)alloc(sizeof(float) * KS[0]);
  S.keys  = (unsigned*)alloc(sizeof(unsigned) * N0);
  S.hist  = (unsigned*)alloc(sizeof(unsigned) * 131072);
  S.sel   = (unsigned*)alloc(sizeof(unsigned) * 8);
  S.prefix = (unsigned long long*)alloc(sizeof(unsigned long long) * N0);
  S.sums   = (unsigned long long*)alloc(sizeof(unsigned long long) * 512);
  S.mapping = (int*)alloc(sizeof(int) * N0);

  int* perm[3];
  perm[0] = (int*)alloc(sizeof(int) * KS[0]);
  perm[1] = (int*)alloc(sizeof(int) * KS[1]);
  perm[2] = (int*)alloc(sizeof(int) * KS[2]);

  const int sizes[4] = {N0, KS[0], KS[1], KS[2]};

  // ---- encoder ----
  run_gcn(x_in, rowL[0], colL[0], wL[0], Wenc[0], benc[0], N0, FIN, CC,
          xlev[0], /*relu=*/1, S, stream);
  for (int i = 1; i <= 3; ++i) {
    run_pool(xlev[i - 1], ppool[i - 1], sizes[i - 1], KS[i - 1],
             rowL[i - 1], colL[i - 1], wL[i - 1],
             rowL[i], colL[i], wL[i], perm[i - 1], tmp, S, stream);
    run_gcn(tmp, rowL[i], colL[i], wL[i], Wenc[i], benc[i], KS[i - 1], CC, CC,
            xlev[i], /*relu=*/1, S, stream);
  }

  // ---- decoder ----
  const float* xcur = xlev[3];   // bottom
  for (int i = 0; i < 3; ++i) {
    const int j   = 2 - i;       // resolution level
    const int n   = sizes[j];
    const int kup = KS[j];       // rows of xcur
    // combined = res ; combined[perm] += xcur
    k_copy_f32<<<nblk((long long)n * CC, 256), 256, 0, stream>>>(tmp, xlev[j],
                                                                 (long long)n * CC);
    k_scatter_add<<<nblk((long long)kup * CC, 256), 256, 0, stream>>>(tmp, xcur,
                                                                      perm[j], kup, CC);
    if (i < 2) {
      run_gcn(tmp, rowL[j], colL[j], wL[j], Wdec[i], bdec[i], n, CC, CC,
              xd, /*relu=*/1, S, stream);
      xcur = xd;
    } else {
      run_gcn(tmp, rowL[0], colL[0], wL[0], Wdec[2], bdec[2], N0, CC, NCLS,
              (float*)d_out, /*relu=*/0, S, stream);
    }
  }
  k_logsoftmax<<<nblk(N0, 256), 256, 0, stream>>>((float*)d_out, N0);
}
